// RuleNetwork_24575802868445
// MI455X (gfx1250) — matrix operand order repaired and checked by reading, not compiled, not run on hardware
//
#include <hip/hip_runtime.h>
#include <hip/hip_bf16.h>
#include <math.h>

// Problem constants
#define B_    512
#define V_    32
#define R_    16
#define D_    512
#define RD_   64
#define H_    4
#define FF_   2048
#define S_    48              // V + R
#define MTOT  (B_ * S_)       // 24576 rows for all big GEMMs

typedef __attribute__((ext_vector_type(16))) _Float16 v16h;
typedef __attribute__((ext_vector_type(8)))  _Float16 v8h;
typedef __attribute__((ext_vector_type(8)))  float    v8f;
typedef __attribute__((ext_vector_type(4)))  float    v4f;

union F16Frag { v16h v; v8h h8[2]; };

#define WMMA_F16(A_, B_m, C_) \
  __builtin_amdgcn_wmma_f32_16x16x32_f16(false, (A_), false, (B_m), (short)0, (C_), false, false)

// Async global->LDS 16B copy (CDNA5, tracked on ASYNCcnt). One inst offset
// applies to BOTH lds and global addresses (ISA §15.18 async pseudocode).
__device__ __forceinline__ void async_ld_lds_b128(const void* gptr, void* lptr) {
  unsigned l = (unsigned)(unsigned long long)lptr;   // low 32b of generic LDS ptr = LDS offset
  asm volatile("global_load_async_to_lds_b128 %0, %1, off" :: "v"(l), "v"(gptr) : "memory");
}
__device__ __forceinline__ void async_ld_lds_b128_off16(const void* gptr, void* lptr) {
  unsigned l = (unsigned)(unsigned long long)lptr;
  asm volatile("global_load_async_to_lds_b128 %0, %1, off offset:16" :: "v"(l), "v"(gptr) : "memory");
}
__device__ __forceinline__ void wait_async0() {
  asm volatile("s_wait_asynccnt 0x0" ::: "memory");
}

// ---- WMMA fragment loaders (CDNA5 ISA §7.12.2 layouts) ----
// A 16x32 f16: lane<16 holds row M=lane, K=0..7 then K=16..23; lane>=16: K=8..15, K=24..31
__device__ __forceinline__ v16h frag_ld_A(const _Float16* base, int ld, int k0, int lane) {
  int r = lane & 15, sel = lane >> 4;
  const _Float16* p = base + r * ld + k0 + sel * 8;
  F16Frag u;
  u.h8[0] = *(const v8h*)(p);
  u.h8[1] = *(const v8h*)(p + 16);
  return u.v;
}
// B 32x16 f16: lane<16 holds col N=lane, K=0..7 then K=16..23; lane>=16: K=8..15, K=24..31.
// Source is row-major [N][K] (weight rows / transposed operand) -> contiguous.
__device__ __forceinline__ v16h frag_ld_B(const _Float16* base, int ld, int k0, int lane) {
  int n = lane & 15, sel = lane >> 4;
  const _Float16* p = base + n * ld + k0 + sel * 8;
  F16Frag u;
  u.h8[0] = *(const v8h*)(p);
  u.h8[1] = *(const v8h*)(p + 16);
  return u.v;
}

// =====================================================================
// Templated tiled GEMM: C[M,N] = act(A[M,K] @ W[N,K]^T + bias [+ res])
// Block = 256 thr (8 waves), block tile 128x128, wave tile 32x64
// (2x4 WMMA frags => 8 WMMA per 32-K step), double-buffered LDS.
// B tile (and A tile when f16) is filled with GLOBAL_LOAD_ASYNC_TO_LDS_B128;
// the f32 A path stages through registers (needs the f32->f16 convert).
// =====================================================================
template <int AF16, int HASRES, int RELU, int O16>
__global__ void __launch_bounds__(256)
gemm_kernel(const float* __restrict__ Af32, const _Float16* __restrict__ Af16,
            const _Float16* __restrict__ W, const float* __restrict__ bias,
            const float* __restrict__ res, float* __restrict__ out32,
            _Float16* __restrict__ out16, int M, int N, int K)
{
  __shared__ _Float16 sA[2][128 * 40];
  __shared__ _Float16 sB[2][128 * 40];
  const int t = threadIdx.x;
  const int lane = t & 31;
  const int wave = t >> 5;
  const int mBlocks = M >> 7;
  const int bm = blockIdx.x % mBlocks;
  const int bn = blockIdx.x / mBlocks;
  const int rowBase = bm * 128, colBase = bn * 128;
  const int wm = (wave & 3) * 32;       // wave grid 4 (M) x 2 (N)
  const int wn = (wave >> 2) * 64;

  // staging thread -> (row, 16-col chunk) for both A (128x32) and B (128x32)
  const int ar = t >> 1, ac = (t & 1) * 16;

  v4f ra[4];     // A staging registers (f32 path only)

  auto asyncB = [&](int buf, int k0) {
    const _Float16* g = W + (size_t)(colBase + ar) * K + k0 + ac;
    _Float16* l = sB[buf] + ar * 40 + ac;
    async_ld_lds_b128(g, l);
    async_ld_lds_b128_off16(g, l);
  };
  auto asyncA = [&](int buf, int k0) {
    const _Float16* g = Af16 + (size_t)(rowBase + ar) * K + k0 + ac;
    _Float16* l = sA[buf] + ar * 40 + ac;
    async_ld_lds_b128(g, l);
    async_ld_lds_b128_off16(g, l);
  };
  auto fetchA32 = [&](int k0) {
    const float* g = Af32 + (size_t)(rowBase + ar) * K + k0 + ac;
    ra[0] = *(const v4f*)(g);
    ra[1] = *(const v4f*)(g + 4);
    ra[2] = *(const v4f*)(g + 8);
    ra[3] = *(const v4f*)(g + 12);
  };
  auto stageA32 = [&](int buf) {
    _Float16* dA = sA[buf] + ar * 40 + ac;
    #pragma unroll
    for (int q = 0; q < 4; ++q) {
      dA[q * 4 + 0] = (_Float16)ra[q][0];
      dA[q * 4 + 1] = (_Float16)ra[q][1];
      dA[q * 4 + 2] = (_Float16)ra[q][2];
      dA[q * 4 + 3] = (_Float16)ra[q][3];
    }
  };

  v8f acc[2][4];
  #pragma unroll
  for (int mi = 0; mi < 2; ++mi)
    #pragma unroll
    for (int ni = 0; ni < 4; ++ni) acc[mi][ni] = (v8f){};

  const int nk = K >> 5;
  asyncB(0, 0);
  if (AF16) {
    asyncA(0, 0);
  } else {
    fetchA32(0);
    stageA32(0);
  }
  wait_async0();
  __syncthreads();

  for (int kt = 0; kt < nk; ++kt) {
    const int cur = kt & 1, nxt = cur ^ 1;
    if (kt + 1 < nk) {                          // async DMA overlaps WMMA below
      const int k1 = (kt + 1) << 5;
      asyncB(nxt, k1);
      if (AF16) asyncA(nxt, k1);
      else      fetchA32(k1);
    }
    if (kt + 2 < nk) {                          // prefetch tile-after-next into caches
      const int k2 = (kt + 2) << 5;
      if (AF16) __builtin_prefetch(Af16 + (size_t)(rowBase + ar) * K + k2 + ac, 0, 1);
      else      __builtin_prefetch(Af32 + (size_t)(rowBase + ar) * K + k2 + ac, 0, 1);
      __builtin_prefetch(W + (size_t)(colBase + ar) * K + k2 + ac, 0, 1);
    }

    const _Float16* bA = sA[cur];
    const _Float16* bB = sB[cur];
    v16h a0 = frag_ld_A(bA + (wm)      * 40, 40, 0, lane);
    v16h a1 = frag_ld_A(bA + (wm + 16) * 40, 40, 0, lane);
    #pragma unroll
    for (int ni = 0; ni < 4; ++ni) {
      v16h bf = frag_ld_B(bB + (wn + ni * 16) * 40, 40, 0, lane);
      acc[0][ni] = WMMA_F16(a0, bf, acc[0][ni]);
      acc[1][ni] = WMMA_F16(a1, bf, acc[1][ni]);
    }
    if (!AF16 && kt + 1 < nk) stageA32(nxt);    // writes the buffer NOT being read
    wait_async0();                              // next tile's DMA landed in LDS
    __syncthreads();
  }

  // epilogue: C frag: lane<16 -> col=lane, rows 0..7; lane>=16 -> col=lane-16, rows 8..15
  const int cn = lane & 15, rsel = (lane >> 4) * 8;
  #pragma unroll
  for (int mi = 0; mi < 2; ++mi) {
    #pragma unroll
    for (int ni = 0; ni < 4; ++ni) {
      const int col = colBase + wn + ni * 16 + cn;
      const float bv = bias[col];
      #pragma unroll
      for (int i = 0; i < 8; ++i) {
        const int row = rowBase + wm + mi * 16 + rsel + i;
        float v = acc[mi][ni][i] + bv;
        if (HASRES) v += res[(size_t)row * N + col];
        if (RELU)   v = v > 0.f ? v : 0.f;
        if (O16) out16[(size_t)row * N + col] = (_Float16)v;
        else     out32[(size_t)row * N + col] = v;
      }
    }
  }
}

// =====================================================================
// Attention core: one block per (batch, head). S=48, hd=128. 128 thr (4 waves).
// S = QK^T (WMMA), softmax in LDS, O = P V (WMMA, V transposed in LDS).
// =====================================================================
__global__ void __launch_bounds__(128)
attn_kernel(const _Float16* __restrict__ qkv, _Float16* __restrict__ attnOut)
{
  __shared__ _Float16 sQ[48 * 136];
  __shared__ _Float16 sK[48 * 136];
  __shared__ _Float16 sVt[128 * 56];
  __shared__ float    sS[48 * 49];
  __shared__ _Float16 sP[48 * 72];
  const int t = threadIdx.x, lane = t & 31, wave = t >> 5;
  const int b = blockIdx.x >> 2, h = blockIdx.x & 3;
  const _Float16* base = qkv + (size_t)b * 48 * 1536 + h * 128;

  // Q/K: vectorized 16B chunks (48 rows x 16 chunks); V: scalar transpose.
  for (int c = t; c < 48 * 16; c += 128) {
    int s = c >> 4, j8 = (c & 15) * 8;
    const _Float16* row = base + (size_t)s * 1536;
    *(v8h*)(sQ + s * 136 + j8) = *(const v8h*)(row + j8);
    *(v8h*)(sK + s * 136 + j8) = *(const v8h*)(row + 512 + j8);
  }
  for (int i = t; i < 48 * 128; i += 128) {
    int s = i >> 7, j = i & 127;
    sVt[j * 56 + s] = base[(size_t)s * 1536 + 1024 + j];
  }
  __syncthreads();

  // scores: 3x3 tiles of 16x16, K=128
  for (int job = wave; job < 9; job += 4) {
    int tm = job / 3, tn = job % 3;
    v8f acc = {};
    #pragma unroll
    for (int kt = 0; kt < 4; ++kt) {
      v16h a = frag_ld_A(sQ + tm * 16 * 136, 136, kt * 32, lane);
      v16h bm = frag_ld_B(sK + tn * 16 * 136, 136, kt * 32, lane);
      acc = WMMA_F16(a, bm, acc);
    }
    int cn = lane & 15, rs = (lane >> 4) * 8;
    #pragma unroll
    for (int i = 0; i < 8; ++i)
      sS[(tm * 16 + rs + i) * 49 + tn * 16 + cn] = acc[i];
  }
  __syncthreads();

  // softmax row-wise (48 rows), write f16 probs padded to K=64
  if (t < 48) {
    const float sc = 0.08838834764831845f;  // 1/sqrt(128)
    float mx = -1e30f;
    for (int j = 0; j < 48; ++j) mx = fmaxf(mx, sS[t * 49 + j] * sc);
    float sum = 0.f;
    for (int j = 0; j < 48; ++j) sum += __expf(sS[t * 49 + j] * sc - mx);
    float inv = 1.f / sum;
    for (int j = 0; j < 48; ++j)
      sP[t * 72 + j] = (_Float16)(__expf(sS[t * 49 + j] * sc - mx) * inv);
    for (int j = 48; j < 64; ++j) sP[t * 72 + j] = (_Float16)0.f;
  }
  __syncthreads();

  // O = P @ V : 3x8 tiles, K=64 (padded)
  for (int job = wave; job < 24; job += 4) {
    int tm = job >> 3, tn = job & 7;
    v8f acc = {};
    #pragma unroll
    for (int kt = 0; kt < 2; ++kt) {
      v16h a = frag_ld_A(sP + tm * 16 * 72, 72, kt * 32, lane);
      v16h bm = frag_ld_B(sVt + tn * 16 * 56, 56, kt * 32, lane);
      acc = WMMA_F16(a, bm, acc);
    }
    int cn = lane & 15, rs = (lane >> 4) * 8;
    #pragma unroll
    for (int i = 0; i < 8; ++i) {
      int s = tm * 16 + rs + i, j = tn * 16 + cn;
      attnOut[((size_t)b * 48 + s) * 512 + h * 128 + j] = (_Float16)acc[i];
    }
  }
}

// =====================================================================
// Final MHA: only head-averaged attention probs. One block per batch.
// =====================================================================
__global__ void __launch_bounds__(128)
attnw_kernel(const _Float16* __restrict__ qk, float* __restrict__ attnw)
{
  __shared__ _Float16 sQ[48 * 136];
  __shared__ _Float16 sK[48 * 136];
  __shared__ float    sS[48 * 49];
  __shared__ float    sAcc[48 * 48];
  const int t = threadIdx.x, lane = t & 31, wave = t >> 5;
  const int b = blockIdx.x;

  for (int i = t; i < 48 * 48; i += 128) sAcc[i] = 0.f;

  for (int h = 0; h < 4; ++h) {
    __syncthreads();
    const _Float16* base = qk + (size_t)b * 48 * 1024 + h * 128;
    for (int c = t; c < 48 * 16; c += 128) {
      int s = c >> 4, j8 = (c & 15) * 8;
      const _Float16* row = base + (size_t)s * 1024;
      *(v8h*)(sQ + s * 136 + j8) = *(const v8h*)(row + j8);
      *(v8h*)(sK + s * 136 + j8) = *(const v8h*)(row + 512 + j8);
    }
    __syncthreads();
    for (int job = wave; job < 9; job += 4) {
      int tm = job / 3, tn = job % 3;
      v8f acc = {};
      #pragma unroll
      for (int kt = 0; kt < 4; ++kt) {
        v16h a = frag_ld_A(sQ + tm * 16 * 136, 136, kt * 32, lane);
        v16h bm = frag_ld_B(sK + tn * 16 * 136, 136, kt * 32, lane);
        acc = WMMA_F16(a, bm, acc);
      }
      int cn = lane & 15, rs = (lane >> 4) * 8;
      #pragma unroll
      for (int i = 0; i < 8; ++i)
        sS[(tm * 16 + rs + i) * 49 + tn * 16 + cn] = acc[i];
    }
    __syncthreads();
    if (t < 48) {
      const float sc = 0.08838834764831845f;
      float mx = -1e30f;
      for (int j = 0; j < 48; ++j) mx = fmaxf(mx, sS[t * 49 + j] * sc);
      float sum = 0.f;
      for (int j = 0; j < 48; ++j) sum += __expf(sS[t * 49 + j] * sc - mx);
      float inv = 1.f / sum;
      for (int j = 0; j < 48; ++j)
        sAcc[t * 48 + j] += __expf(sS[t * 49 + j] * sc - mx) * inv;
    }
  }
  __syncthreads();
  for (int i = t; i < 48 * 48; i += 128) attnw[(size_t)b * 2304 + i] = sAcc[i] * 0.25f;
}

// ---- LayerNorm: one block (256 thr) per 512-wide row ----
__global__ void __launch_bounds__(256)
ln_kernel(const float* __restrict__ in, const float* __restrict__ w,
          const float* __restrict__ bsh, float* __restrict__ out)
{
  __shared__ float red[256];
  const int r = blockIdx.x, t = threadIdx.x;
  const float* x = in + (size_t)r * 512;
  float a = x[t], c = x[t + 256];
  red[t] = a + c;
  __syncthreads();
  for (int s = 128; s > 0; s >>= 1) { if (t < s) red[t] += red[t + s]; __syncthreads(); }
  float mean = red[0] * (1.f / 512.f);
  __syncthreads();
  float da = a - mean, dc = c - mean;
  red[t] = da * da + dc * dc;
  __syncthreads();
  for (int s = 128; s > 0; s >>= 1) { if (t < s) red[t] += red[t + s]; __syncthreads(); }
  float inv = rsqrtf(red[0] * (1.f / 512.f) + 1e-5f);
  float* o = out + (size_t)r * 512;
  o[t]       = da * inv * w[t]       + bsh[t];
  o[t + 256] = dc * inv * w[t + 256] + bsh[t + 256];
}

// ---- small elementwise / prep kernels ----
__global__ void f32_to_f16_kernel(const float* __restrict__ s, _Float16* __restrict__ d, int n) {
  int i = blockIdx.x * 256 + threadIdx.x;
  if (i < n) d[i] = (_Float16)s[i];
}

__global__ void pe_kernel(float* __restrict__ pe) {
  int i = blockIdx.x * 256 + threadIdx.x;
  if (i >= S_ * D_) return;
  int s = i / D_, d = i % D_;
  float div = __expf(-(float)(d & ~1) * (logf(10000.f) / (float)D_));
  float v = (float)s * div;
  pe[i] = (d & 1) ? cosf(v) : sinf(v);
}

__global__ void ruleproj_kernel(const float* __restrict__ rule_emb, const float* __restrict__ W_rule,
                                const float* __restrict__ b_rule, float* __restrict__ rp) {
  int i = blockIdx.x * 256 + threadIdx.x;
  if (i >= R_ * D_) return;
  int r = i / D_, d = i % D_;
  float acc = b_rule[d];
  for (int j = 0; j < RD_; ++j) acc += rule_emb[r * RD_ + j] * W_rule[d * RD_ + j];
  rp[i] = acc;
}

__global__ void x0_kernel(const float* __restrict__ hidden, const float* __restrict__ rp,
                          const float* __restrict__ pe, float* __restrict__ x) {
  int i = blockIdx.x * 256 + threadIdx.x;  // over B*48*512
  if (i >= MTOT * D_) return;
  int d = i % D_;
  int bs = i / D_;
  int s = bs % S_, b = bs / S_;
  float v = (s < V_) ? hidden[((size_t)b * V_ + s) * D_ + d] : rp[(s - V_) * D_ + d];
  x[i] = v + pe[s * D_ + d];
}

__global__ void scores_kernel(const float* __restrict__ attnw, float* __restrict__ scores) {
  int i = blockIdx.x * 256 + threadIdx.x;  // over B*V*R
  if (i >= B_ * V_ * R_) return;
  int b = i / (V_ * R_);
  int rem = i % (V_ * R_);
  int v = rem / R_, r = rem % R_;
  const float* A = attnw + (size_t)b * S_ * S_;
  scores[i] = A[v * S_ + (V_ + r)] + A[(V_ + r) * S_ + v];
}

// straight-through Gumbel mask == one-hot(argmax(scores + g)); scatter one row.
__global__ void __launch_bounds__(256)
finalize_kernel(const float* __restrict__ scores, const float* __restrict__ gu,
                const float* __restrict__ ruleProj, float* __restrict__ out)
{
  __shared__ float bv[256];
  __shared__ int   bi[256];
  const int b = blockIdx.x, t = threadIdx.x;
  const float* sc = scores + (size_t)b * (V_ * R_);
  const float* u  = gu + (size_t)b * (V_ * R_);
  float v1 = sc[t]       - logf(-logf(u[t]));
  float v2 = sc[t + 256] - logf(-logf(u[t + 256]));
  float best = v1; int bidx = t;
  if (v2 > best) { best = v2; bidx = t + 256; }
  bv[t] = best; bi[t] = bidx;
  __syncthreads();
  for (int s = 128; s > 0; s >>= 1) {
    if (t < s) {
      if (bv[t + s] > bv[t] || (bv[t + s] == bv[t] && bi[t + s] < bi[t])) {
        bv[t] = bv[t + s]; bi[t] = bi[t + s];
      }
    }
    __syncthreads();
  }
  const int idx = bi[0];
  const int vstar = idx >> 4, rstar = idx & 15;
  const float coef = sc[idx];
  float* ob = out + (size_t)b * V_ * D_;
  for (int j = t; j < V_ * D_; j += 256) ob[j] = 0.f;
  __syncthreads();
  for (int d = t; d < D_; d += 256) ob[vstar * D_ + d] = coef * ruleProj[rstar * D_ + d];
}

// =====================================================================
// Host-side orchestration
// =====================================================================
extern "C" void kernel_launch(void* const* d_in, const int* in_sizes, int n_in,
                              void* d_out, int out_size, void* d_ws, size_t ws_size,
                              hipStream_t stream) {
  (void)in_sizes; (void)n_in; (void)out_size; (void)ws_size;

  const float* hidden   = (const float*)d_in[0];
  const float* gumbel_u = (const float*)d_in[1];
  const float* rule_emb = (const float*)d_in[2];
  const float* W_rule   = (const float*)d_in[3];
  const float* b_rule   = (const float*)d_in[4];
  struct Layer { const float *Wqkv,*bqkv,*Wo,*bo,*W1,*b1,*W2,*b2,*ln1w,*ln1b,*ln2w,*ln2b; };
  Layer L[3];
  int li = 5;
  for (int p = 0; p < 3; ++p) {
    L[p].Wqkv=(const float*)d_in[li+0];  L[p].bqkv=(const float*)d_in[li+1];
    L[p].Wo  =(const float*)d_in[li+2];  L[p].bo  =(const float*)d_in[li+3];
    L[p].W1  =(const float*)d_in[li+4];  L[p].b1  =(const float*)d_in[li+5];
    L[p].W2  =(const float*)d_in[li+6];  L[p].b2  =(const float*)d_in[li+7];
    L[p].ln1w=(const float*)d_in[li+8];  L[p].ln1b=(const float*)d_in[li+9];
    L[p].ln2w=(const float*)d_in[li+10]; L[p].ln2b=(const float*)d_in[li+11];
    li += 12;
  }
  const float* mhaWqkv = (const float*)d_in[41];
  const float* mhabqkv = (const float*)d_in[42];

  // workspace carve-up
  char* ws = (char*)d_ws;
  size_t off = 0;
  auto alloc = [&](size_t bytes) -> void* {
    off = (off + 255) & ~(size_t)255;
    void* p = ws + off;
    off += bytes;
    return p;
  };
  float*    pe      = (float*)alloc((size_t)S_ * D_ * 4);
  float*    ruleP   = (float*)alloc((size_t)R_ * D_ * 4);
  float*    xf32    = (float*)alloc((size_t)MTOT * D_ * 4);
  float*    tmp     = (float*)alloc((size_t)MTOT * D_ * 4);
  _Float16* big     = (_Float16*)alloc((size_t)MTOT * FF_ * 2);  // qkv | h1 | qk overlays
  _Float16* qkv     = big;                                        // [M,1536]
  _Float16* attnBuf = big + (size_t)MTOT * 1536;                  // [M,512] (after qkv)
  _Float16* h1      = big;                                        // [M,2048]
  _Float16* qkBuf   = big;                                        // [M,1024]
  float*    attnw   = (float*)alloc((size_t)B_ * S_ * S_ * 4);
  float*    scores  = (float*)alloc((size_t)B_ * V_ * R_ * 4);
  _Float16* wQkv[3], *wW1[3], *wW2[3], *wO[3];
  for (int p = 0; p < 3; ++p) {
    wQkv[p] = (_Float16*)alloc((size_t)1536 * 512 * 2);
    wW1[p]  = (_Float16*)alloc((size_t)2048 * 512 * 2);
    wW2[p]  = (_Float16*)alloc((size_t)512 * 2048 * 2);
    wO[p]   = (_Float16*)alloc((size_t)512 * 512 * 2);
  }
  _Float16* wMha = (_Float16*)alloc((size_t)1536 * 512 * 2);

  auto cvt = [&](const float* s, _Float16* d, int n) {
    f32_to_f16_kernel<<<(n + 255) / 256, 256, 0, stream>>>(s, d, n);
  };
  for (int p = 0; p < 3; ++p) {
    cvt(L[p].Wqkv, wQkv[p], 1536 * 512);
    cvt(L[p].Wo,   wO[p],   512 * 512);
    cvt(L[p].W1,   wW1[p],  2048 * 512);
    cvt(L[p].W2,   wW2[p],  512 * 2048);
  }
  cvt(mhaWqkv, wMha, 1536 * 512);

  pe_kernel<<<(S_ * D_ + 255) / 256, 256, 0, stream>>>(pe);
  ruleproj_kernel<<<(R_ * D_ + 255) / 256, 256, 0, stream>>>(rule_emb, W_rule, b_rule, ruleP);
  x0_kernel<<<(MTOT * D_ + 255) / 256, 256, 0, stream>>>(hidden, ruleP, pe, xf32);

  // Three used GEMM modes, compile-time specialized:
  auto gemm_a32_o16 = [&](const float* A32, const _Float16* Wm, const float* bias,
                          _Float16* o16, int N, int K, int relu) {
    dim3 grid((MTOT / 128) * (N / 128));
    if (relu)
      gemm_kernel<0, 0, 1, 1><<<grid, 256, 0, stream>>>(A32, nullptr, Wm, bias, nullptr,
                                                        nullptr, o16, MTOT, N, K);
    else
      gemm_kernel<0, 0, 0, 1><<<grid, 256, 0, stream>>>(A32, nullptr, Wm, bias, nullptr,
                                                        nullptr, o16, MTOT, N, K);
  };
  auto gemm_a16_res_o32 = [&](const _Float16* A16, const _Float16* Wm, const float* bias,
                              const float* res, float* o32, int N, int K) {
    dim3 grid((MTOT / 128) * (N / 128));
    gemm_kernel<1, 1, 0, 0><<<grid, 256, 0, stream>>>(nullptr, A16, Wm, bias, res,
                                                      o32, nullptr, MTOT, N, K);
  };

  for (int p = 0; p < 3; ++p) {
    // QKV
    gemm_a32_o16(xf32, wQkv[p], L[p].bqkv, qkv, 1536, 512, 0);
    // SDPA per (b, h)
    attn_kernel<<<B_ * H_, 128, 0, stream>>>(qkv, attnBuf);
    // O-proj + residual
    gemm_a16_res_o32(attnBuf, wO[p], L[p].bo, xf32, tmp, 512, 512);
    ln_kernel<<<MTOT, 256, 0, stream>>>(tmp, L[p].ln1w, L[p].ln1b, xf32);
    // FF1 (ReLU)
    gemm_a32_o16(xf32, wW1[p], L[p].b1, h1, 2048, 512, 1);
    // FF2 + residual
    gemm_a16_res_o32(h1, wW2[p], L[p].b2, xf32, tmp, 512, 2048);
    ln_kernel<<<MTOT, 256, 0, stream>>>(tmp, L[p].ln2w, L[p].ln2b, xf32);
  }

  // Final MHA: only Q,K (first 1024 output features of Wqkv) are needed.
  gemm_a32_o16(xf32, wMha, mhabqkv, qkBuf, 1024, 512, 0);
  attnw_kernel<<<B_, 128, 0, stream>>>(qkBuf, attnw);
  scores_kernel<<<(B_ * V_ * R_ + 255) / 256, 256, 0, stream>>>(attnw, scores);
  finalize_kernel<<<B_, 256, 0, stream>>>(scores, gumbel_u, ruleP, (float*)d_out);
}
